// Attention_54451595378942
// MI455X (gfx1250) — matrix operand order repaired and checked by reading, not compiled, not run on hardware
//
#include <hip/hip_runtime.h>
#include <hip/hip_bf16.h>

typedef __attribute__((ext_vector_type(16))) _Float16 v16h;
typedef __attribute__((ext_vector_type(8)))  _Float16 h8v;
typedef __attribute__((ext_vector_type(8)))  float    v8f;

#define B_   32
#define C_   192
#define H_   56
#define W_   56
#define HW_  3136
#define M_   100352      // B*H*W
#define QKV_ 576
#define HEADS_ 6
#define WIN_ 7
#define N_   49
#define NW_  64          // windows per image
#define DHEAD_ 32

// ---------------- fragment loaders ----------------
// A fragment 16x32 f16: lane r=l&15 -> row; K halves = 16*hi + 0..15
__device__ __forceinline__ v16h load_a_frag(const _Float16* base, int rowBase, int stride, int kBase) {
    int lane = threadIdx.x & 31;
    int r = lane & 15, hi = lane >> 4;
    const _Float16* p = base + (size_t)(rowBase + r) * stride + kBase + hi * 16;
    h8v lo = *(const h8v*)p;
    h8v hh = *(const h8v*)(p + 8);
    v16h f;
#pragma unroll
    for (int j = 0; j < 8; ++j) { f[j] = lo[j]; f[j + 8] = hh[j]; }
    return f;
}

// B fragment 32x16 f16 loaded from B^T stored row-major [N, K]:
// lane r -> column n; K halves = 16*hi + 0..15 (one contiguous 32B chunk)
__device__ __forceinline__ v16h load_b_frag(const _Float16* baseT, int colBase, int stride, int kBase) {
    int lane = threadIdx.x & 31;
    int r = lane & 15, hi = lane >> 4;
    const _Float16* p = baseT + (size_t)(colBase + r) * stride + kBase + hi * 16;
    return *(const v16h*)p;
}

// ---------------- prep: weight f16 conversion + rel-pos bias ----------------
__global__ void prep_kernel(const float* __restrict__ pw_w, const float* __restrict__ proj_w,
                            const float* __restrict__ table,
                            _Float16* __restrict__ pw16, _Float16* __restrict__ pj16,
                            float* __restrict__ biasf) {
    int t = blockIdx.x * blockDim.x + threadIdx.x;
    if (t < QKV_ * C_) pw16[t] = (_Float16)pw_w[t];
    if (t < C_ * C_)   pj16[t] = (_Float16)proj_w[t];
    if (t < HEADS_ * N_ * N_) {
        int hd = t / (N_ * N_);
        int rem = t - hd * (N_ * N_);
        int i = rem / N_, j = rem - i * N_;
        int r1 = i / WIN_, c1 = i % WIN_, r2 = j / WIN_, c2 = j % WIN_;
        int idx = (r1 - r2 + WIN_ - 1) * (2 * WIN_ - 1) + (c1 - c2 + WIN_ - 1);
        biasf[t] = table[idx * HEADS_ + hd];
    }
}

// ---------------- depthwise 3x3 + BN -> f16 NHWC ----------------
__global__ void dwbn_kernel(const float* __restrict__ x, const float* __restrict__ dw,
                            const float* __restrict__ g, const float* __restrict__ bt,
                            const float* __restrict__ mn, const float* __restrict__ vr,
                            _Float16* __restrict__ y16) {
    int n = blockIdx.x * 256 + threadIdx.x;
    if (n >= B_ * C_ * HW_) return;
    int w = n % W_;
    int h = (n / W_) % H_;
    int c = (n / HW_) % C_;
    int b = n / (HW_ * C_);
    const float* xb = x + (size_t)(b * C_ + c) * HW_;
    const float* wk = dw + c * 9;
    float acc = 0.f;
#pragma unroll
    for (int dy = -1; dy <= 1; ++dy) {
        int hh = h + dy;
        if (hh < 0 || hh >= H_) continue;
#pragma unroll
        for (int dx = -1; dx <= 1; ++dx) {
            int ww = w + dx;
            if (ww < 0 || ww >= W_) continue;
            acc += wk[(dy + 1) * 3 + (dx + 1)] * xb[hh * W_ + ww];
        }
    }
    float inv = g[c] * rsqrtf(vr[c] + 1e-5f);
    float yv = (acc - mn[c]) * inv + bt[c];
    y16[(size_t)(b * HW_ + h * W_ + w) * C_ + c] = (_Float16)yv;
}

// ---------------- qkv pointwise GEMM: [M,192] x [192,576] + bias -> f16 ----------------
__global__ void __launch_bounds__(128) qkv_gemm(const _Float16* __restrict__ y16,
                                                const _Float16* __restrict__ pw16,
                                                const float* __restrict__ pw_b,
                                                _Float16* __restrict__ qkv16) {
    int wave = threadIdx.x >> 5;
    int mBase = blockIdx.x * 16;
    int nBase = (blockIdx.y * 4 + wave) * 16;
    int lane = threadIdx.x & 31, r = lane & 15, hi = lane >> 4;
    v8f acc = {};
#pragma unroll
    for (int kb = 0; kb < 6; ++kb) {
        v16h a = load_a_frag(y16, mBase, C_, kb * 32);
        v16h b = load_b_frag(pw16, nBase, C_, kb * 32);
        acc = __builtin_amdgcn_wmma_f32_16x16x32_f16(false, a, false, b, (short)0, acc, false, false);
    }
    int o = nBase + r;
    float bias = pw_b[o];
#pragma unroll
    for (int i = 0; i < 8; ++i) {
        int m = mBase + hi * 8 + i;
        qkv16[(size_t)m * QKV_ + o] = (_Float16)(acc[i] + bias);
    }
}

// ---------------- per-(window, head) attention ----------------
__global__ void __launch_bounds__(128) attn_kernel(const _Float16* __restrict__ qkv16,
                                                   const float* __restrict__ mask,
                                                   const float* __restrict__ biasf,
                                                   _Float16* __restrict__ ao16) {
    __shared__ __align__(32) _Float16 qs[64 * 32];
    __shared__ __align__(32) _Float16 ksm[64 * 32];
    __shared__ __align__(32) _Float16 vT[32 * 64];
    __shared__ __align__(32) _Float16 Ps[64 * 64];
    __shared__ float Ss[64 * 68];

    int tid = threadIdx.x;
    int wIdx = blockIdx.x;       // 0..2047  (b*64 + win)
    int head = blockIdx.y;       // 0..5
    int b = wIdx >> 6;
    int win = wIdx & 63;
    int sxi = win >> 3, syi = win & 7;

    // zero padded LDS tiles
    for (int t = tid; t < 1024; t += 128) {
        ((unsigned*)qs)[t] = 0u; ((unsigned*)ksm)[t] = 0u; ((unsigned*)vT)[t] = 0u;
    }
    for (int t = tid; t < 2048; t += 128) ((unsigned*)Ps)[t] = 0u;
    __syncthreads();

    // load q,k rows [j][d], v transposed [d][j]
    for (int t = tid; t < N_ * 4; t += 128) {
        int j = t >> 2, ch = t & 3;
        int wy = j / WIN_, wx = j % WIN_;
        int m = b * HW_ + (sxi * WIN_ + wy) * W_ + (syi * WIN_ + wx);
        const _Float16* src = qkv16 + (size_t)m * QKV_ + head * DHEAD_ + ch * 8;
        h8v qv = *(const h8v*)(src);
        h8v kv = *(const h8v*)(src + C_);
        h8v vv = *(const h8v*)(src + 2 * C_);
        *(h8v*)(qs + j * 32 + ch * 8)  = qv;
        *(h8v*)(ksm + j * 32 + ch * 8) = kv;
#pragma unroll
        for (int u = 0; u < 8; ++u) vT[(ch * 8 + u) * 64 + j] = vv[u];
    }
    __syncthreads();

    int wave = tid >> 5;
    int lane = tid & 31, r = lane & 15, hi = lane >> 4;

    // S = q * k^T   (each wave: 16-row block, 4 column tiles, K=32 in one WMMA)
    {
        v16h a = load_a_frag(qs, wave * 16, 32, 0);
#pragma unroll
        for (int tj = 0; tj < 4; ++tj) {
            v16h bf = load_b_frag(ksm, tj * 16, 32, 0);
            v8f acc = {};
            acc = __builtin_amdgcn_wmma_f32_16x16x32_f16(false, a, false, bf, (short)0, acc, false, false);
#pragma unroll
            for (int i = 0; i < 8; ++i)
                Ss[(wave * 16 + hi * 8 + i) * 68 + tj * 16 + r] = acc[i];
        }
    }
    __syncthreads();

    // softmax over valid 49 columns (scale + rel bias + mask), write P f16
    const float scale = 0.07216878364870322f;   // 192^-0.5
    if (tid < N_) {
        int i = tid;
        const float* bia = biasf + (head * N_ + i) * N_;
        const float* msk = mask + (win * N_ + i) * N_;
        float* srow = &Ss[i * 68];
        float mx = -1e30f;
        for (int j = 0; j < N_; ++j) {
            float v = srow[j] * scale + bia[j] + msk[j];
            srow[j] = v;
            mx = fmaxf(mx, v);
        }
        float sum = 0.f;
        for (int j = 0; j < N_; ++j) sum += __expf(srow[j] - mx);
        float is = 1.f / sum;
        for (int j = 0; j < N_; ++j)
            Ps[i * 64 + j] = (_Float16)(__expf(srow[j] - mx) * is);
    }
    __syncthreads();

    // out = P * v   (each wave: 16-row block, 2 d-tiles, K=64 in two WMMAs)
#pragma unroll
    for (int tj = 0; tj < 2; ++tj) {
        v8f acc = {};
#pragma unroll
        for (int kb = 0; kb < 2; ++kb) {
            v16h a  = load_a_frag(Ps, wave * 16, 64, kb * 32);
            v16h bf = load_b_frag(vT, tj * 16, 64, kb * 32);
            acc = __builtin_amdgcn_wmma_f32_16x16x32_f16(false, a, false, bf, (short)0, acc, false, false);
        }
#pragma unroll
        for (int i = 0; i < 8; ++i) {
            int jrow = wave * 16 + hi * 8 + i;
            if (jrow < N_) {
                int wy = jrow / WIN_, wx = jrow % WIN_;
                int m = b * HW_ + (sxi * WIN_ + wy) * W_ + (syi * WIN_ + wx);
                ao16[(size_t)m * C_ + head * DHEAD_ + tj * 16 + r] = (_Float16)acc[i];
            }
        }
    }
}

// ---------------- projection GEMM: [M,192] x [192,192] + bias -> fp32 NCHW ----------------
__global__ void __launch_bounds__(128) proj_gemm(const _Float16* __restrict__ ao16,
                                                 const _Float16* __restrict__ pj16,
                                                 const float* __restrict__ proj_b,
                                                 float* __restrict__ out) {
    int wave = threadIdx.x >> 5;
    int mBase = blockIdx.x * 16;
    int nBase = (blockIdx.y * 4 + wave) * 16;
    int lane = threadIdx.x & 31, r = lane & 15, hi = lane >> 4;
    v8f acc = {};
#pragma unroll
    for (int kb = 0; kb < 6; ++kb) {
        v16h a = load_a_frag(ao16, mBase, C_, kb * 32);
        v16h b = load_b_frag(pj16, nBase, C_, kb * 32);
        acc = __builtin_amdgcn_wmma_f32_16x16x32_f16(false, a, false, b, (short)0, acc, false, false);
    }
    int o = nBase + r;
    float bias = proj_b[o];
#pragma unroll
    for (int i = 0; i < 8; ++i) {
        int m = mBase + hi * 8 + i;
        int bb = m / HW_, rem = m % HW_;
        out[((size_t)bb * C_ + o) * HW_ + rem] = acc[i] + bias;
    }
}

// ---------------- launch ----------------
extern "C" void kernel_launch(void* const* d_in, const int* in_sizes, int n_in,
                              void* d_out, int out_size, void* d_ws, size_t ws_size,
                              hipStream_t stream) {
    const float* x      = (const float*)d_in[0];
    const float* mask   = (const float*)d_in[1];
    const float* dw_w   = (const float*)d_in[2];
    const float* bn_g   = (const float*)d_in[3];
    const float* bn_b   = (const float*)d_in[4];
    const float* bn_m   = (const float*)d_in[5];
    const float* bn_v   = (const float*)d_in[6];
    const float* pw_w   = (const float*)d_in[7];
    const float* pw_b   = (const float*)d_in[8];
    const float* proj_w = (const float*)d_in[9];
    const float* proj_b = (const float*)d_in[10];
    const float* table  = (const float*)d_in[11];

    char* ws = (char*)d_ws;
    const size_t offQKV  = 0;                                   // 100352*576*2 = 115,605,504
    const size_t offY    = offQKV + (size_t)M_ * QKV_ * 2;      // y f16 / attn-out f16 (reused)
    const size_t offPW   = offY + (size_t)M_ * C_ * 2;          // 576*192 f16
    const size_t offPJ   = offPW + (size_t)QKV_ * C_ * 2;       // 192*192 f16
    const size_t offBias = offPJ + (size_t)C_ * C_ * 2;         // 6*49*49 f32

    _Float16* qkv16 = (_Float16*)(ws + offQKV);
    _Float16* y16   = (_Float16*)(ws + offY);    // also reused as attention output
    _Float16* pw16  = (_Float16*)(ws + offPW);
    _Float16* pj16  = (_Float16*)(ws + offPJ);
    float*    biasf = (float*)(ws + offBias);
    float*    out   = (float*)d_out;

    prep_kernel<<<dim3((QKV_ * C_ + 255) / 256), dim3(256), 0, stream>>>(pw_w, proj_w, table, pw16, pj16, biasf);
    dwbn_kernel<<<dim3((B_ * C_ * HW_ + 255) / 256), dim3(256), 0, stream>>>(x, dw_w, bn_g, bn_b, bn_m, bn_v, y16);
    qkv_gemm<<<dim3(M_ / 16, QKV_ / 64), dim3(128), 0, stream>>>(y16, pw16, pw_b, qkv16);
    attn_kernel<<<dim3(B_ * NW_, HEADS_), dim3(128), 0, stream>>>(qkv16, mask, biasf, y16);
    proj_gemm<<<dim3(M_ / 16, C_ / 64), dim3(128), 0, stream>>>(y16, pj16, proj_b, out);
}
